// MultiHeadAttention_20229295964210
// MI455X (gfx1250) — matrix operand order repaired and checked by reading, not compiled, not run on hardware
//
#include <hip/hip_runtime.h>

// Problem sizes fixed by the reference
#define B_  4
#define S_  1024
#define D_  1024
#define H_  16
#define HD_ 64
#define T_  2048   // 2*S

typedef __attribute__((ext_vector_type(16))) __bf16 v16bf;
typedef __attribute__((ext_vector_type(2)))  __bf16 v2bf;
typedef __attribute__((ext_vector_type(8)))  float  v8f;

typedef __attribute__((ext_vector_type(4))) unsigned int u32x4;
typedef __attribute__((ext_vector_type(4))) int          i32x4;
typedef __attribute__((ext_vector_type(8))) int          i32x8;

#if __has_builtin(__builtin_amdgcn_tensor_load_to_lds) && \
    __has_builtin(__builtin_amdgcn_s_wait_tensorcnt)
#define USE_TDM 1
#else
#define USE_TDM 0
#endif

// ---- compile-probe diagnostics (show up in stderr; free signal) ----
#if USE_TDM
#pragma message "CDNA5 probe: tensor_load_to_lds PRESENT (TDM path enabled)"
#if defined(__clang_major__) && (__clang_major__ >= 23)
#pragma message "CDNA5 probe: using 6-arg tensor_load_to_lds (clang >= 23)"
#else
#pragma message "CDNA5 probe: using 5-arg tensor_load_to_lds (clang < 23)"
#endif
#else
#pragma message "CDNA5 probe: tensor_load_to_lds ABSENT (vector staging fallback)"
#endif
#if __has_builtin(__builtin_amdgcn_cvt_pk_bf16_f32)
#pragma message "CDNA5 probe: cvt_pk_bf16_f32 builtin PRESENT"
#else
#pragma message "CDNA5 probe: cvt_pk_bf16_f32 builtin ABSENT (__bf16 cast path)"
#endif

union Frag16 {
  v16bf bf;
  unsigned int u[8];
};

__device__ __forceinline__ unsigned short f2bf(float f) {
  __bf16 b = (__bf16)f;                 // native f32->bf16 convert
  unsigned short u;
  __builtin_memcpy(&u, &b, 2);
  return u;
}

// Packed f32x2 -> bf16x2 (hardware v_cvt_pk_bf16_f32 when available)
__device__ __forceinline__ unsigned int pack_bf16(float lo, float hi) {
#if __has_builtin(__builtin_amdgcn_cvt_pk_bf16_f32)
  auto r = __builtin_amdgcn_cvt_pk_bf16_f32(lo, hi);
  unsigned int u;
  __builtin_memcpy(&u, &r, 4);
  return u;
#else
  v2bf v;
  v[0] = (__bf16)lo;
  v[1] = (__bf16)hi;
  unsigned int u;
  __builtin_memcpy(&u, &v, 4);
  return u;
#endif
}

__device__ __forceinline__ float fexp2(float x) {
#if __has_builtin(__builtin_amdgcn_exp2f)
  return __builtin_amdgcn_exp2f(x);
#else
  return exp2f(x);
#endif
}

// LDS byte offset of a __shared__ object (generic addr low 32 bits = LDS offset)
__device__ __forceinline__ unsigned lds_off(const void* p) {
  return (unsigned)(size_t)p;
}

#if USE_TDM
// Issue a 2-D Tensor-Data-Mover copy: tileH rows x tileW elements (2-byte
// elements) from global (row stride = rowStride elements) into LDS at ldsOff,
// packed contiguously. One issue per wave moves the whole tile; completion
// tracked by TENSORcnt (in-order per wave -> usable for double buffering).
__device__ __forceinline__ void tdm_load_2d_bf16(unsigned ldsOff,
                                                 const void* tileStart,
                                                 unsigned tileW, unsigned tileH,
                                                 unsigned tensorW, unsigned tensorH,
                                                 unsigned rowStride) {
  const unsigned long long ga = (unsigned long long)(size_t)tileStart;
  u32x4 g0;
  g0[0] = 1u;                                     // count=1, user descriptor
  g0[1] = ldsOff;                                 // lds_addr (bytes)
  g0[2] = (unsigned)(ga & 0xFFFFFFFFu);           // global_addr[31:0]
  g0[3] = (unsigned)((ga >> 32) & 0x1FFFFFFu)     // global_addr[56:32]
          | (2u << 30);                           // type=2 ("image")
  i32x8 g1;
  g1[0] = (int)(1u << 16);                        // data_size=1 (2 bytes)
  g1[1] = (int)((tensorW & 0xFFFFu) << 16);       // tensor_dim0[15:0]
  g1[2] = (int)(((tensorW >> 16) & 0xFFFFu) |     // tensor_dim0[31:16]
                ((tensorH & 0xFFFFu) << 16));     // tensor_dim1[15:0]
  g1[3] = (int)(((tensorH >> 16) & 0xFFFFu) |     // tensor_dim1[31:16]
                ((tileW & 0xFFFFu) << 16));       // tile_dim0
  g1[4] = (int)(tileH & 0xFFFFu);                 // tile_dim1 (tile_dim2 = 0)
  g1[5] = (int)rowStride;                         // tensor_dim0_stride[31:0]
  g1[6] = 0;
  g1[7] = 0;
  i32x4 z4 = {0, 0, 0, 0};
#if defined(__clang_major__) && (__clang_major__ >= 23)
  i32x8 z8 = {0, 0, 0, 0, 0, 0, 0, 0};
  __builtin_amdgcn_tensor_load_to_lds(g0, g1, z4, z4, z8, 0);
#else
  __builtin_amdgcn_tensor_load_to_lds(g0, g1, z4, z4, 0);
#endif
}
#endif // USE_TDM

// A fragment: 16x32 bf16 (M x K). Lane l<16 holds row M=l (K chunks 0-7,16-23),
// lane l+16 holds row M=l (K chunks 8-15,24-31). L is row-major [m][k].
__device__ __forceinline__ Frag16 load_afrag(const unsigned short* L, int stride,
                                             int m0, int k0, int lane) {
  Frag16 f;
  const int r = lane & 15, hf = lane >> 4;
  const unsigned short* p = L + (size_t)(m0 + r) * stride + k0 + 8 * hf;
  #pragma unroll
  for (int i = 0; i < 8; ++i) {
    const int k = 2 * i + ((i >= 4) ? 8 : 0);
    f.u[i] = *(const unsigned int*)(p + k);
  }
  return f;
}

// B fragment: 32x16 bf16 (K x N). Lanes 0-15 hold K=0-7,16-23, lanes 16-31 K=8-15,24-31,
// lane column = lane&15. L is the TRANSPOSE of B, i.e. L[n][k] == B[k][n].
__device__ __forceinline__ Frag16 load_bfrag(const unsigned short* L, int stride,
                                             int n0, int k0, int lane) {
  Frag16 f;
  const int n = lane & 15, hf = lane >> 4;
  const unsigned short* p = L + (size_t)(n0 + n) * stride + k0 + 8 * hf;
  #pragma unroll
  for (int i = 0; i < 8; ++i)
    f.u[i] = *(const unsigned int*)(p + 2 * i + ((i >= 4) ? 8 : 0));
  return f;
}

__device__ __forceinline__ v8f wmma_bf16(const Frag16& a, const Frag16& b, v8f c) {
  return __builtin_amdgcn_wmma_f32_16x16x32_bf16(false, a.bf, false, b.bf,
                                                 (short)0, c, false, false);
}

// ---------------------------------------------------------------------------
// fp32 -> bf16 pairwise conversion (for weights)
// ---------------------------------------------------------------------------
__global__ void k_cvt_bf16(const float* __restrict__ src,
                           unsigned short* __restrict__ dst, int nPairs) {
  int i = blockIdx.x * blockDim.x + threadIdx.x;
  const int stride = gridDim.x * blockDim.x;
  for (; i < nPairs; i += stride) {
    const float2 f = ((const float2*)src)[i];
    ((unsigned int*)dst)[i] = pack_bf16(f.x, f.y);
  }
}

// ---------------------------------------------------------------------------
// LayerNorm over D=1024, one block (8 waves) per row, packed bf16 output
// ---------------------------------------------------------------------------
__global__ __launch_bounds__(256) void k_layernorm_bf16(
    const float* __restrict__ x, const float* __restrict__ gamma,
    const float* __restrict__ beta, unsigned short* __restrict__ out) {
  const int row = blockIdx.x;
  const float* xp = x + (size_t)row * D_;
  const int c0 = 2 * threadIdx.x;       // adjacent pair #1
  const int c1 = 2 * threadIdx.x + 512; // adjacent pair #2
  float v[4], s = 0.f, s2 = 0.f;
  v[0] = xp[c0]; v[1] = xp[c0 + 1]; v[2] = xp[c1]; v[3] = xp[c1 + 1];
  #pragma unroll
  for (int k = 0; k < 4; ++k) { s += v[k]; s2 += v[k] * v[k]; }
  #pragma unroll
  for (int m = 1; m < 32; m <<= 1) {
    s  += __shfl_xor(s,  m, 32);
    s2 += __shfl_xor(s2, m, 32);
  }
  __shared__ float red[2][8];
  const int wid = threadIdx.x >> 5, lane = threadIdx.x & 31;
  if (lane == 0) { red[0][wid] = s; red[1][wid] = s2; }
  __syncthreads();
  if (threadIdx.x == 0) {
    float a = 0.f, c = 0.f;
    #pragma unroll
    for (int w = 0; w < 8; ++w) { a += red[0][w]; c += red[1][w]; }
    red[0][0] = a; red[1][0] = c;
  }
  __syncthreads();
  const float mu  = red[0][0] * (1.0f / D_);
  const float var = red[1][0] * (1.0f / D_) - mu * mu;
  const float inv = rsqrtf(var + 1e-5f);
  unsigned short* op = out + (size_t)row * D_;
  const float y0 = (v[0] - mu) * inv * gamma[c0]     + beta[c0];
  const float y1 = (v[1] - mu) * inv * gamma[c0 + 1] + beta[c0 + 1];
  const float y2 = (v[2] - mu) * inv * gamma[c1]     + beta[c1];
  const float y3 = (v[3] - mu) * inv * gamma[c1 + 1] + beta[c1 + 1];
  *(unsigned int*)&op[c0] = pack_bf16(y0, y1);
  *(unsigned int*)&op[c1] = pack_bf16(y2, y3);
}

// ---------------------------------------------------------------------------
// GEMM: Y[m,n] = sum_k A[m,k] * W[n,k]   (y = x @ W.T), K = N = 1024
// Block tile 128x128, 8 waves (4x2), each wave 32x64 via 2x4 WMMA tiles.
// Double-buffered TDM pipeline: wave 0 issues the k+1 slab copies while all
// waves compute on slab k (TENSORcnt completions are in-order per wave).
// ---------------------------------------------------------------------------
__global__ __launch_bounds__(256) void k_gemm_bf16(
    const void* __restrict__ Ain, int aIsF32,
    const unsigned short* __restrict__ W,
    void* __restrict__ Out, int outIsBf16,
    long aBatchStride, long outBatchStride, long outOffset) {
  const int K = D_, N = D_;
  const int NK = K / 32;
  const int bz = blockIdx.z;
  const int mBase = blockIdx.y * 128;
  const int nBase = blockIdx.x * 128;
  const int t = threadIdx.x;
  const int lane = t & 31, wid = t >> 5;
  const int waveM = wid >> 1, waveN = wid & 1;

  __shared__ __align__(16) unsigned short sA[2][128 * 32];
  __shared__ __align__(16) unsigned short sW[2][128 * 32];

  const float*          Af = (const float*)Ain          + (size_t)bz * aBatchStride;
  const unsigned short* Ab = (const unsigned short*)Ain + (size_t)bz * aBatchStride;

  v8f acc[2][4];
  #pragma unroll
  for (int mt = 0; mt < 2; ++mt)
    #pragma unroll
    for (int nt = 0; nt < 4; ++nt)
      #pragma unroll
      for (int i = 0; i < 8; ++i) acc[mt][nt][i] = 0.f;

  const int r  = t >> 1;         // 0..127: A row staged by this thread (f32 path)
  const int ch = (t & 1) * 16;   // which 16-element half of the 32-wide K slab

#if USE_TDM
  if (t < 32) {   // prologue: issue slab 0
    if (!aIsF32)
      tdm_load_2d_bf16(lds_off(&sA[0][0]), Ab + (size_t)mBase * K,
                       32, 128, (unsigned)K, 128, (unsigned)K);
    tdm_load_2d_bf16(lds_off(&sW[0][0]), W + (size_t)nBase * K,
                     32, 128, (unsigned)K, 128, (unsigned)K);
  }
#endif

  for (int kb = 0; kb < NK; ++kb) {
    const int cur = kb & 1, nxt = cur ^ 1;
#if USE_TDM
    if (t < 32) __builtin_amdgcn_s_wait_tensorcnt(0);
    __syncthreads();   // (a) slab kb visible; previous compute finished
    if (kb + 1 < NK && t < 32) {   // overlap DMA of slab kb+1 with compute kb
      if (!aIsF32)
        tdm_load_2d_bf16(lds_off(&sA[nxt][0]),
                         Ab + (size_t)mBase * K + (kb + 1) * 32,
                         32, 128, (unsigned)K, 128, (unsigned)K);
      tdm_load_2d_bf16(lds_off(&sW[nxt][0]),
                       W + (size_t)nBase * K + (kb + 1) * 32,
                       32, 128, (unsigned)K, 128, (unsigned)K);
    }
    if (aIsF32) {      // kv GEMM: threads stage A with fused f32->bf16 packing
      const float* ap = Af + (size_t)(mBase + r) * K + kb * 32 + ch;
      unsigned int w[8];
      #pragma unroll
      for (int x = 0; x < 8; ++x)
        w[x] = pack_bf16(ap[2 * x], ap[2 * x + 1]);
      *(uint4*)&sA[cur][r * 32 + ch]     = make_uint4(w[0], w[1], w[2], w[3]);
      *(uint4*)&sA[cur][r * 32 + ch + 8] = make_uint4(w[4], w[5], w[6], w[7]);
      if (kb + 1 < NK) __builtin_prefetch(ap + 32, 0, 1);
      __syncthreads(); // (b) thread-staged A visible
    }
#else
    __syncthreads();
    if (aIsF32) {
      const float* ap = Af + (size_t)(mBase + r) * K + kb * 32 + ch;
      unsigned int w[8];
      #pragma unroll
      for (int x = 0; x < 8; ++x)
        w[x] = pack_bf16(ap[2 * x], ap[2 * x + 1]);
      *(uint4*)&sA[cur][r * 32 + ch]     = make_uint4(w[0], w[1], w[2], w[3]);
      *(uint4*)&sA[cur][r * 32 + ch + 8] = make_uint4(w[4], w[5], w[6], w[7]);
      if (kb + 1 < NK) __builtin_prefetch(ap + 32, 0, 1);
    } else {
      const unsigned short* ap = Ab + (size_t)(mBase + r) * K + kb * 32 + ch;
      *(uint4*)&sA[cur][r * 32 + ch]     = ((const uint4*)ap)[0];
      *(uint4*)&sA[cur][r * 32 + ch + 8] = ((const uint4*)ap)[1];
      if (kb + 1 < NK) __builtin_prefetch(ap + 32, 0, 1);
    }
    {
      const unsigned short* wp = W + (size_t)(nBase + r) * K + kb * 32 + ch;
      *(uint4*)&sW[cur][r * 32 + ch]     = ((const uint4*)wp)[0];
      *(uint4*)&sW[cur][r * 32 + ch + 8] = ((const uint4*)wp)[1];
      if (kb + 1 < NK) __builtin_prefetch(wp + 32, 0, 1);
    }
    __syncthreads();
#endif

    Frag16 af[2], bfr[4];
    #pragma unroll
    for (int mt = 0; mt < 2; ++mt)
      af[mt] = load_afrag(sA[cur], 32, waveM * 32 + mt * 16, 0, lane);
    #pragma unroll
    for (int nt = 0; nt < 4; ++nt)
      bfr[nt] = load_bfrag(sW[cur], 32, waveN * 64 + nt * 16, 0, lane);
    #pragma unroll
    for (int mt = 0; mt < 2; ++mt)
      #pragma unroll
      for (int nt = 0; nt < 4; ++nt)
        acc[mt][nt] = wmma_bf16(af[mt], bfr[nt], acc[mt][nt]);
  }

  // epilogue (C layout: lane&15 = column, vgpr i = row i (+8 for upper half))
  const int cl = lane & 15, hf = lane >> 4;
  #pragma unroll
  for (int mt = 0; mt < 2; ++mt)
    #pragma unroll
    for (int nt = 0; nt < 4; ++nt) {
      const int col = nBase + waveN * 64 + nt * 16 + cl;
      #pragma unroll
      for (int i = 0; i < 8; ++i) {
        const int row = mBase + waveM * 32 + mt * 16 + i + 8 * hf;
        const size_t oi = (size_t)bz * outBatchStride + outOffset +
                          (size_t)row * N + col;
        const float val = acc[mt][nt][i];
        if (outIsBf16) ((unsigned short*)Out)[oi] = f2bf(val);
        else           ((float*)Out)[oi]          = val;
      }
    }
}

// ---------------------------------------------------------------------------
// Flash attention: out = softmax(Q K^T / 8) V + residual, K == V (bf16 kv).
// Block = 8 waves x 16 query rows = 128 rows of one (b,h). Key tiles streamed
// through a double-buffered TDM pipeline; transposed copy built for PV.
// ---------------------------------------------------------------------------
__global__ __launch_bounds__(256) void k_flash_attn(
    const float* __restrict__ Qf, const unsigned short* __restrict__ kv,
    float* __restrict__ out) {
  const int bz = blockIdx.z, h = blockIdx.y;
  const int qBase = blockIdx.x * 128;
  const int t = threadIdx.x;
  const int lane = t & 31, wid = t >> 5;
  const int NB = T_ / 64;

  __shared__ __align__(16) unsigned short sK [2][64 * 64];
  __shared__ __align__(16) unsigned short sKt[64 * 64];
  __shared__ __align__(16) unsigned short sP [8 * 16 * 64];
  unsigned short* myP = sP + wid * (16 * 64);

  const unsigned short* kvBase = kv + (size_t)bz * T_ * D_ + h * HD_;

  // Q fragments (16 rows x 64 dims), fp32 -> bf16, resident in registers
  Frag16 qf[2];
  {
    const int rr = lane & 15, hf = lane >> 4;
    const float* qp = Qf + ((size_t)bz * S_ + qBase + wid * 16 + rr) * D_ + h * HD_;
    #pragma unroll
    for (int ks = 0; ks < 2; ++ks)
      #pragma unroll
      for (int i = 0; i < 8; ++i) {
        const int k = ks * 32 + 2 * i + ((i >= 4) ? 8 : 0) + 8 * hf;
        qf[ks].u[i] = pack_bf16(qp[k], qp[k + 1]);
      }
  }

  v8f o[4];
  float mrow[8], lrow[8];
  #pragma unroll
  for (int nt = 0; nt < 4; ++nt)
    #pragma unroll
    for (int i = 0; i < 8; ++i) o[nt][i] = 0.f;
  #pragma unroll
  for (int i = 0; i < 8; ++i) { mrow[i] = -3.0e38f; lrow[i] = 0.f; }

  const float scale = 0.125f;        // 1/sqrt(64)
  const float L2E   = 1.44269504f;

#if USE_TDM
  if (t < 32)   // prologue: issue key tile 0
    tdm_load_2d_bf16(lds_off(&sK[0][0]), kvBase, 64, 64, D_, 64, D_);
#endif

  for (int jb = 0; jb < NB; ++jb) {
    const int cur = jb & 1, nxt = cur ^ 1;
#if USE_TDM
    if (t < 32) __builtin_amdgcn_s_wait_tensorcnt(0);
    __syncthreads();   // (a) tile jb visible; previous compute finished
    if (jb + 1 < NB && t < 32)   // overlap DMA of tile jb+1 with compute jb
      tdm_load_2d_bf16(lds_off(&sK[nxt][0]),
                       kvBase + (size_t)(jb + 1) * 64 * D_, 64, 64, D_, 64, D_);
    {  // build transposed copy for the PV pass
      const int j = t >> 2, q4 = (t & 3) * 16;
      uint4 v0 = *(const uint4*)&sK[cur][j * 64 + q4];
      uint4 v1 = *(const uint4*)&sK[cur][j * 64 + q4 + 8];
      const unsigned short* e0 = (const unsigned short*)&v0;
      const unsigned short* e1 = (const unsigned short*)&v1;
      #pragma unroll
      for (int x = 0; x < 8; ++x) sKt[(q4 + x) * 64 + j]     = e0[x];
      #pragma unroll
      for (int x = 0; x < 8; ++x) sKt[(q4 + 8 + x) * 64 + j] = e1[x];
    }
#else
    __syncthreads();
    {  // stage 64 keys x 64 dims, both layouts (K == V)
      const int j = t >> 2, q4 = (t & 3) * 16;
      const unsigned short* src = kvBase + (size_t)(jb * 64 + j) * D_ + q4;
      uint4 v0 = ((const uint4*)src)[0];
      uint4 v1 = ((const uint4*)src)[1];
      *(uint4*)&sK[cur][j * 64 + q4]     = v0;
      *(uint4*)&sK[cur][j * 64 + q4 + 8] = v1;
      const unsigned short* e0 = (const unsigned short*)&v0;
      const unsigned short* e1 = (const unsigned short*)&v1;
      #pragma unroll
      for (int x = 0; x < 8; ++x) sKt[(q4 + x) * 64 + j]     = e0[x];
      #pragma unroll
      for (int x = 0; x < 8; ++x) sKt[(q4 + 8 + x) * 64 + j] = e1[x];
    }
#endif
    __syncthreads();   // (b) sKt (and sK) ready

    // scores: 16 q-rows x 64 keys (B frag L = sK since B[d][j] = K[j][d])
    v8f s[4];
    #pragma unroll
    for (int nt = 0; nt < 4; ++nt)
      #pragma unroll
      for (int i = 0; i < 8; ++i) s[nt][i] = 0.f;
    #pragma unroll
    for (int ks = 0; ks < 2; ++ks)
      #pragma unroll
      for (int nt = 0; nt < 4; ++nt)
        s[nt] = wmma_bf16(qf[ks], load_bfrag(sK[cur], 64, nt * 16, ks * 32, lane),
                          s[nt]);

    // online softmax (row reductions stay within 16-lane halves)
    #pragma unroll
    for (int i = 0; i < 8; ++i) {
      float mx = fmaxf(fmaxf(s[0][i], s[1][i]), fmaxf(s[2][i], s[3][i])) * scale;
      #pragma unroll
      for (int m = 1; m <= 8; m <<= 1) mx = fmaxf(mx, __shfl_xor(mx, m, 32));
      const float nm   = fmaxf(mrow[i], mx);
      const float corr = fexp2((mrow[i] - nm) * L2E);
      mrow[i] = nm;
      float rs = 0.f;
      const int row = i + 8 * (lane >> 4);
      const int cl  = lane & 15;
      #pragma unroll
      for (int nt = 0; nt < 4; ++nt) {
        const float p = fexp2((s[nt][i] * scale - nm) * L2E);
        rs += p;
        myP[row * 64 + nt * 16 + cl] = f2bf(p);   // C-layout -> LDS (A-layout src)
      }
      #pragma unroll
      for (int m = 1; m <= 8; m <<= 1) rs += __shfl_xor(rs, m, 32);
      lrow[i] = lrow[i] * corr + rs;
      #pragma unroll
      for (int nt = 0; nt < 4; ++nt) o[nt][i] *= corr;
    }

    // O += P @ V   (B frag L = sKt since B[j][d] = V[j][d] = sKt[d][j]^T)
    Frag16 pf0 = load_afrag(myP, 64, 0, 0,  lane);
    Frag16 pf1 = load_afrag(myP, 64, 0, 32, lane);
    #pragma unroll
    for (int nt = 0; nt < 4; ++nt) {
      o[nt] = wmma_bf16(pf0, load_bfrag(sKt, 64, nt * 16, 0,  lane), o[nt]);
      o[nt] = wmma_bf16(pf1, load_bfrag(sKt, 64, nt * 16, 32, lane), o[nt]);
    }
  }

  // epilogue: out = O / l + residual (residual is the fp32 Q projection)
  const int cl = lane & 15, hf = lane >> 4;
  const size_t rowBase = (size_t)bz * S_ + qBase + wid * 16;
  #pragma unroll
  for (int i = 0; i < 8; ++i) {
    const int row   = i + 8 * hf;
    const float inv = 1.0f / lrow[i];
    const float* rp = Qf  + (rowBase + row) * D_ + h * HD_;
    float*       op = out + (rowBase + row) * D_ + h * HD_;
    #pragma unroll
    for (int nt = 0; nt < 4; ++nt) {
      const int d = nt * 16 + cl;
      op[d] = o[nt][i] * inv + rp[d];
    }
  }
}

// ---------------------------------------------------------------------------
extern "C" void kernel_launch(void* const* d_in, const int* in_sizes, int n_in,
                              void* d_out, int out_size, void* d_ws, size_t ws_size,
                              hipStream_t stream) {
  (void)in_sizes; (void)n_in; (void)out_size; (void)ws_size;

  const float* x1  = (const float*)d_in[0];
  const float* x2  = (const float*)d_in[1];
  const float* g1  = (const float*)d_in[2];
  const float* be1 = (const float*)d_in[3];
  const float* g2  = (const float*)d_in[4];
  const float* be2 = (const float*)d_in[5];
  const float* Wq1 = (const float*)d_in[6];
  const float* Wq2 = (const float*)d_in[7];
  const float* Wkv = (const float*)d_in[8];

  char* ws = (char*)d_ws;
  const size_t MB = 1024 * 1024;
  unsigned short* xh1  = (unsigned short*)(ws + 0);        //  8 MB  LN(x1) bf16
  unsigned short* xh2  = (unsigned short*)(ws + 8  * MB);  //  8 MB  LN(x2) bf16
  unsigned short* wq1h = (unsigned short*)(ws + 16 * MB);  //  2 MB
  unsigned short* wq2h = (unsigned short*)(ws + 18 * MB);  //  2 MB
  unsigned short* wkvh = (unsigned short*)(ws + 20 * MB);  //  2 MB
  float*          q11  = (float*)        (ws + 22 * MB);   // 16 MB  Q11 f32
  float*          q22  = (float*)        (ws + 38 * MB);   // 16 MB  Q22 f32
  unsigned short* kvh  = (unsigned short*)(ws + 54 * MB);  // 16 MB  kv bf16 [B,2S,D]

  // 1) weights -> bf16 (pairwise packed conversion)
  k_cvt_bf16<<<1024, 256, 0, stream>>>(Wq1, wq1h, D_ * D_ / 2);
  k_cvt_bf16<<<1024, 256, 0, stream>>>(Wq2, wq2h, D_ * D_ / 2);
  k_cvt_bf16<<<1024, 256, 0, stream>>>(Wkv, wkvh, D_ * D_ / 2);

  // 2) layernorm -> bf16 activations
  k_layernorm_bf16<<<B_ * S_, 256, 0, stream>>>(x1, g1, be1, xh1);
  k_layernorm_bf16<<<B_ * S_, 256, 0, stream>>>(x2, g2, be2, xh2);

  // 3) projections (WMMA bf16, f32 accumulate)
  dim3 gg(D_ / 128, S_ / 128, B_);
  // Q11 = LN(x1) @ Wq1.T  (f32 out)
  k_gemm_bf16<<<gg, 256, 0, stream>>>(xh1, 0, wq1h, q11, 0,
                                      (long)S_ * D_, (long)S_ * D_, 0);
  // Q22 = LN(x2) @ Wq2.T  (f32 out)
  k_gemm_bf16<<<gg, 256, 0, stream>>>(xh2, 0, wq2h, q22, 0,
                                      (long)S_ * D_, (long)S_ * D_, 0);
  // kv = concat(Q11,Q22) @ Wkv.T  (bf16 out; concat realized via row offset)
  k_gemm_bf16<<<gg, 256, 0, stream>>>(q11, 1, wkvh, kvh, 1,
                                      (long)S_ * D_, (long)T_ * D_, 0);
  k_gemm_bf16<<<gg, 256, 0, stream>>>(q22, 1, wkvh, kvh, 1,
                                      (long)S_ * D_, (long)T_ * D_, (long)S_ * D_);

  // 4) attention + residual
  dim3 ga(S_ / 128, H_, B_);
  float* out1 = (float*)d_out;
  float* out2 = out1 + (size_t)B_ * S_ * D_;
  k_flash_attn<<<ga, 256, 0, stream>>>(q11, kvh, out1);
  k_flash_attn<<<ga, 256, 0, stream>>>(q22, kvh, out2);
}
